// TransformerLayer_45629732552755
// MI455X (gfx1250) — compile-verified, matrix operand order checked by reading, not run on hardware
//
#include <hip/hip_runtime.h>
#include <hip/hip_bf16.h>
#include <math.h>

// ---------------------------------------------------------------------------
// Types for CDNA5 WMMA (wave32): V_WMMA_F32_16X16X32_BF16
// ---------------------------------------------------------------------------
typedef __bf16 v16bf __attribute__((ext_vector_type(16)));
typedef float  v8f   __attribute__((ext_vector_type(8)));

union FragU {
    uint4 u[2];   // 32 bytes
    v16bf v;      // 16 x bf16 = 32 bytes
};

#define TS 4096   // sequence length S
#define TD 1024   // model dim D
#define TH 2048   // internal dim H

#define BM 128
#define BN 128
#define BK 32
#define TILE_BYTES (BM * BK * 2)   // 8192 bytes per A/B tile

// ---------------------------------------------------------------------------
// Elementwise fp32 -> bf16
// ---------------------------------------------------------------------------
__global__ void tl_cvt_bf16(const float* __restrict__ in, __bf16* __restrict__ out, size_t n) {
    size_t i = (size_t)blockIdx.x * blockDim.x + threadIdx.x;
    if (i < n) out[i] = (__bf16)in[i];
}

// W [K,N] fp32 (row-major)  ->  Wt [N,K] bf16 (row-major)  (i.e. B^T for GEMM)
__global__ void tl_transpose_cvt(const float* __restrict__ W, __bf16* __restrict__ Wt,
                                 int K, int N) {
    size_t i = (size_t)blockIdx.x * blockDim.x + threadIdx.x;
    size_t total = (size_t)K * N;
    if (i >= total) return;
    int n = (int)(i / K);
    int k = (int)(i % K);
    Wt[i] = (__bf16)W[(size_t)k * N + n];
}

// bf16 transpose: in [R,C] -> out [C,R]
__global__ void tl_transpose_b16(const __bf16* __restrict__ in, __bf16* __restrict__ out,
                                 int R, int C) {
    size_t i = (size_t)blockIdx.x * blockDim.x + threadIdx.x;
    size_t total = (size_t)R * C;
    if (i >= total) return;
    int c = (int)(i / R);   // output row
    int r = (int)(i % R);   // output col
    out[i] = in[(size_t)r * C + c];
}

// ---------------------------------------------------------------------------
// Tiled WMMA GEMM:  C[M,N] = scale * (A[M,K] @ Bt[N,K]^T) (+bias)(+resid)(relu)
// A, Bt bf16 row-major. 256 threads = 8 waves; block tile 128x128, BK=32.
//
// Data movement (CDNA5 path): double-buffered LDS tiles filled with
// GLOBAL_LOAD_ASYNC_TO_LDS_B128 (ASYNCcnt), overlapped with WMMA compute on
// the other buffer. s_wait_asynccnt 0 + workgroup barrier publishes a tile.
//
// Fragment LDS loads follow ISA Sec 7.12.2 16-bit layout:
//   lane l: idx = l&15, half = l>>4; K-chunks [8h..8h+7] and [16+8h..16+8h+7].
// ---------------------------------------------------------------------------
__global__ __launch_bounds__(256) void tl_gemm_wmma(
    const __bf16* __restrict__ A, const __bf16* __restrict__ Bt,
    const float* __restrict__ bias, const float* __restrict__ residIn,
    float* __restrict__ outF, __bf16* __restrict__ outB,
    int M, int N, int K, float scale, int doRelu)
{
    __shared__ __align__(16) __bf16 As[2][BM * BK];
    __shared__ __align__(16) __bf16 Bs[2][BN * BK];

    const int tid  = threadIdx.x;
    const int lane = tid & 31;
    const int wid  = tid >> 5;
    const int wm   = wid >> 1;   // 0..3 -> row group of 32
    const int wn   = wid & 1;    // 0..1 -> col group of 64
    const int half = lane >> 4;  // K-chunk select
    const int l16  = lane & 15;

    const int rowBase = blockIdx.y * BM;
    const int colBase = blockIdx.x * BN;

    // LDS byte addresses: generic shared pointer's low 32 bits are the LDS
    // byte offset (shared aperture mapping).
    const uint32_t aBase = (uint32_t)(uintptr_t)(&As[0][0]);
    const uint32_t bBase = (uint32_t)(uintptr_t)(&Bs[0][0]);

    // This thread's two 16B staging slots (s in 0..511 covers a 128x32 tile).
    const int s0 = tid, s1 = tid + 256;
    const int r0 = s0 >> 2, c0 = s0 & 3;
    const int r1 = s1 >> 2, c1 = s1 & 3;
    const __bf16* gA0 = A  + (size_t)(rowBase + r0) * K + 8 * c0;
    const __bf16* gA1 = A  + (size_t)(rowBase + r1) * K + 8 * c1;
    const __bf16* gB0 = Bt + (size_t)(colBase + r0) * K + 8 * c0;
    const __bf16* gB1 = Bt + (size_t)(colBase + r1) * K + 8 * c1;
    const uint32_t la0 = aBase + (uint32_t)s0 * 16, la1 = aBase + (uint32_t)s1 * 16;
    const uint32_t lb0 = bBase + (uint32_t)s0 * 16, lb1 = bBase + (uint32_t)s1 * 16;

    // Issue one tile fill (4 async b128 loads / thread) into buffer `buf`.
    auto issue = [&](int buf, int kb) {
        const uint32_t o = (uint32_t)buf * TILE_BYTES;
        asm volatile("global_load_async_to_lds_b128 %0, %1, off"
                     :: "v"(la0 + o), "v"(gA0 + kb) : "memory");
        asm volatile("global_load_async_to_lds_b128 %0, %1, off"
                     :: "v"(la1 + o), "v"(gA1 + kb) : "memory");
        asm volatile("global_load_async_to_lds_b128 %0, %1, off"
                     :: "v"(lb0 + o), "v"(gB0 + kb) : "memory");
        asm volatile("global_load_async_to_lds_b128 %0, %1, off"
                     :: "v"(lb1 + o), "v"(gB1 + kb) : "memory");
        // speculative prefetch one more tile ahead (global_prefetch_b8)
        __builtin_prefetch(gA0 + kb + BK, 0, 0);
        __builtin_prefetch(gB0 + kb + BK, 0, 0);
    };

    v8f acc[2][4] = {};

    const int nsteps = K / BK;
    issue(0, 0);

    for (int step = 0; step < nsteps; ++step) {
        const int buf = step & 1;

        // Publish buffer `buf`: my async writes done, then all threads'.
        asm volatile("s_wait_asynccnt 0x0" ::: "memory");
        __syncthreads();

        // Overlap: start filling the other buffer while computing this one.
        if (step + 1 < nsteps) issue(buf ^ 1, (step + 1) * BK);

        const __bf16* Asb = &As[buf][0];
        const __bf16* Bsb = &Bs[buf][0];

        // ---- load fragments ----
        v16bf af[2];
#pragma unroll
        for (int i = 0; i < 2; ++i) {
            int r = 32 * wm + 16 * i + l16;
            FragU u;
            u.u[0] = *reinterpret_cast<const uint4*>(&Asb[r * BK + 8 * half]);
            u.u[1] = *reinterpret_cast<const uint4*>(&Asb[r * BK + 16 + 8 * half]);
            af[i] = u.v;
        }
        v16bf bfr[4];
#pragma unroll
        for (int j = 0; j < 4; ++j) {
            int r = 64 * wn + 16 * j + l16;
            FragU u;
            u.u[0] = *reinterpret_cast<const uint4*>(&Bsb[r * BK + 8 * half]);
            u.u[1] = *reinterpret_cast<const uint4*>(&Bsb[r * BK + 16 + 8 * half]);
            bfr[j] = u.v;
        }

        // ---- 8 WMMAs per K-step ----
#pragma unroll
        for (int i = 0; i < 2; ++i)
#pragma unroll
            for (int j = 0; j < 4; ++j)
                acc[i][j] = __builtin_amdgcn_wmma_f32_16x16x32_bf16(
                    false, af[i], false, bfr[j], (short)0, acc[i][j], false, false);
        // No end-of-loop barrier needed: the next iteration's top barrier
        // orders all waves' reads of `buf` before any refill of `buf`.
    }

    // ---- epilogue: C/D layout: VGPR r holds M=r (lanes 0-15), M=8+r (16-31) ----
#pragma unroll
    for (int i = 0; i < 2; ++i) {
#pragma unroll
        for (int j = 0; j < 4; ++j) {
#pragma unroll
            for (int r = 0; r < 8; ++r) {
                int row = rowBase + 32 * wm + 16 * i + r + (half << 3);
                int col = colBase + 64 * wn + 16 * j + l16;
                float v = acc[i][j][r] * scale;
                if (bias)    v += bias[col];
                if (residIn) v += residIn[(size_t)row * N + col];
                if (doRelu)  v = fmaxf(v, 0.0f);
                size_t o = (size_t)row * N + col;
                if (outF) outF[o] = v;
                if (outB) outB[o] = (__bf16)v;
            }
        }
    }
}

// ---------------------------------------------------------------------------
// Causal row softmax on scores (fp32, pre-scaled), in place; also emit bf16.
// One 256-thread block per row.
// ---------------------------------------------------------------------------
__global__ __launch_bounds__(256) void tl_softmax_causal(
    float* __restrict__ attn, __bf16* __restrict__ attnb, int Sdim)
{
    const int row = blockIdx.x;
    const int tid = threadIdx.x;
    const int nvalid = row + 1;
    float* rp = attn + (size_t)row * Sdim;
    __shared__ float red[256];

    float m = -__builtin_inff();
    for (int j = tid; j < nvalid; j += 256) m = fmaxf(m, rp[j]);
    red[tid] = m; __syncthreads();
    for (int s = 128; s > 0; s >>= 1) {
        if (tid < s) red[tid] = fmaxf(red[tid], red[tid + s]);
        __syncthreads();
    }
    m = red[0]; __syncthreads();

    float sum = 0.0f;
    for (int j = tid; j < nvalid; j += 256) {
        float e = __expf(rp[j] - m);
        rp[j] = e;
        sum += e;
    }
    red[tid] = sum; __syncthreads();
    for (int s = 128; s > 0; s >>= 1) {
        if (tid < s) red[tid] += red[tid + s];
        __syncthreads();
    }
    float inv = 1.0f / red[0];

    __bf16* rb = attnb + (size_t)row * Sdim;
    for (int j = tid; j < Sdim; j += 256) {
        float v = (j < nvalid) ? rp[j] * inv : 0.0f;
        rp[j] = v;
        rb[j] = (__bf16)v;
    }
}

// ---------------------------------------------------------------------------
// Launcher
// ---------------------------------------------------------------------------
extern "C" void kernel_launch(void* const* d_in, const int* in_sizes, int n_in,
                              void* d_out, int out_size, void* d_ws, size_t ws_size,
                              hipStream_t stream) {
    (void)in_sizes; (void)n_in; (void)out_size; (void)ws_size;

    const float* x  = (const float*)d_in[0];
    const float* Wq = (const float*)d_in[1];  const float* bq = (const float*)d_in[2];
    const float* Wk = (const float*)d_in[3];  const float* bk = (const float*)d_in[4];
    const float* Wv = (const float*)d_in[5];  const float* bv = (const float*)d_in[6];
    const float* Wo = (const float*)d_in[7];  const float* bo = (const float*)d_in[8];
    const float* W1 = (const float*)d_in[9];  const float* b1 = (const float*)d_in[10];
    const float* W2 = (const float*)d_in[11]; const float* b2 = (const float*)d_in[12];

    float* outMain = (float*)d_out;                       // [S,D]
    float* attnF   = (float*)d_out + (size_t)TS * TD;     // [S,S]

    // workspace layout
    char* base = (char*)d_ws;
    size_t off = 0;
    auto alloc = [&](size_t bytes) -> void* {
        void* p = base + off;
        off += (bytes + 255) & ~(size_t)255;
        return p;
    };
    __bf16* xb     = (__bf16*)alloc((size_t)TS * TD * 2);
    __bf16* wqt    = (__bf16*)alloc((size_t)TH * TD * 2);
    __bf16* wkt    = (__bf16*)alloc((size_t)TH * TD * 2);
    __bf16* wvt    = (__bf16*)alloc((size_t)TH * TD * 2);
    __bf16* wot    = (__bf16*)alloc((size_t)TD * TH * 2);
    __bf16* w1t    = (__bf16*)alloc((size_t)TH * TD * 2);
    __bf16* w2t    = (__bf16*)alloc((size_t)TD * TH * 2);
    __bf16* qb     = (__bf16*)alloc((size_t)TS * TH * 2);
    __bf16* kb     = (__bf16*)alloc((size_t)TS * TH * 2);
    __bf16* vb     = (__bf16*)alloc((size_t)TS * TH * 2);
    __bf16* vbt    = (__bf16*)alloc((size_t)TH * TS * 2);
    __bf16* attnB  = (__bf16*)alloc((size_t)TS * TS * 2);
    __bf16* aob    = (__bf16*)alloc((size_t)TS * TH * 2);
    float*  resid  = (float*) alloc((size_t)TS * TD * 4);
    __bf16* residb = (__bf16*)alloc((size_t)TS * TD * 2);
    __bf16* ffb    = (__bf16*)alloc((size_t)TS * TH * 2);

    const int EB = 256;
    auto eg = [](size_t n) { return (int)((n + 255) / 256); };

    // ---- precision conversion / weight transposes ----
    tl_cvt_bf16<<<eg((size_t)TS * TD), EB, 0, stream>>>(x, xb, (size_t)TS * TD);
    tl_transpose_cvt<<<eg((size_t)TD * TH), EB, 0, stream>>>(Wq, wqt, TD, TH);
    tl_transpose_cvt<<<eg((size_t)TD * TH), EB, 0, stream>>>(Wk, wkt, TD, TH);
    tl_transpose_cvt<<<eg((size_t)TD * TH), EB, 0, stream>>>(Wv, wvt, TD, TH);
    tl_transpose_cvt<<<eg((size_t)TH * TD), EB, 0, stream>>>(Wo, wot, TH, TD);
    tl_transpose_cvt<<<eg((size_t)TD * TH), EB, 0, stream>>>(W1, w1t, TD, TH);
    tl_transpose_cvt<<<eg((size_t)TH * TD), EB, 0, stream>>>(W2, w2t, TH, TD);

    const float inv_sqrt_h = 1.0f / sqrtf((float)TH);

    // ---- Q, K, V projections: [S,D] @ [D,H] + b -> bf16 ----
    dim3 gQKV(TH / BN, TS / BM);
    tl_gemm_wmma<<<gQKV, 256, 0, stream>>>(xb, wqt, bq, nullptr, nullptr, qb,
                                           TS, TH, TD, 1.0f, 0);
    tl_gemm_wmma<<<gQKV, 256, 0, stream>>>(xb, wkt, bk, nullptr, nullptr, kb,
                                           TS, TH, TD, 1.0f, 0);
    tl_gemm_wmma<<<gQKV, 256, 0, stream>>>(xb, wvt, bv, nullptr, nullptr, vb,
                                           TS, TH, TD, 1.0f, 0);

    // ---- scores = q @ k^T * 1/sqrt(H) -> fp32 directly into d_out attn region
    //      (Bt for q@k^T is exactly kb [S,H] row-major) ----
    dim3 gSc(TS / BN, TS / BM);
    tl_gemm_wmma<<<gSc, 256, 0, stream>>>(qb, kb, nullptr, nullptr, attnF, nullptr,
                                          TS, TS, TH, inv_sqrt_h, 0);

    // ---- causal softmax in place; also bf16 copy for AV GEMM ----
    tl_softmax_causal<<<TS, 256, 0, stream>>>(attnF, attnB, TS);

    // ---- v^T for AV GEMM ----
    tl_transpose_b16<<<eg((size_t)TS * TH), EB, 0, stream>>>(vb, vbt, TS, TH);

    // ---- attn_out = attn @ v : [S,S] @ [S,H] ----
    dim3 gAV(TH / BN, TS / BM);
    tl_gemm_wmma<<<gAV, 256, 0, stream>>>(attnB, vbt, nullptr, nullptr, nullptr, aob,
                                          TS, TH, TS, 1.0f, 0);

    // ---- resid = x + attn_out @ Wo + bo  (fp32 + bf16) ----
    dim3 gO(TD / BN, TS / BM);
    tl_gemm_wmma<<<gO, 256, 0, stream>>>(aob, wot, bo, x, resid, residb,
                                         TS, TD, TH, 1.0f, 0);

    // ---- ff1 = relu(resid @ W1 + b1) -> bf16 ----
    dim3 gF1(TH / BN, TS / BM);
    tl_gemm_wmma<<<gF1, 256, 0, stream>>>(residb, w1t, b1, nullptr, nullptr, ffb,
                                          TS, TH, TD, 1.0f, 1);

    // ---- out = resid + ff1 @ W2 + b2 -> fp32 d_out ----
    dim3 gF2(TD / BN, TS / BM);
    tl_gemm_wmma<<<gF2, 256, 0, stream>>>(ffb, w2t, b2, resid, outMain, nullptr,
                                          TS, TD, TH, 1.0f, 0);
}
